// MPJRDLayer_63995012710856
// MI455X (gfx1250) — compile-verified, matrix operand order checked by reading, not run on hardware
//
#include <hip/hip_runtime.h>
#include <stdint.h>

// Problem constants (from reference setup_inputs)
#define B_   128
#define N_   1024
#define K_   512          // D*S = 8*64
#define KC   64           // k-chunk staged per double-buffer step (2 WMMA k=32 steps)
#define ROWF 68           // padded LDS row stride in floats (64 + 4 pad)
#define NCH  (K_ / KC)    // 8 chunks

typedef __attribute__((ext_vector_type(16))) _Float16 v16h;
typedef __attribute__((ext_vector_type(8)))  float    v8f;

// s_wait_asynccnt <imm>: wait until ASYNCcnt <= Nw. "memory" clobber keeps the
// compiler from hoisting the subsequent ds_load reads above the wait.
template <int Nw>
__device__ inline void wait_asynccnt() {
  asm volatile("s_wait_asynccnt %0" ::"n"(Nw) : "memory");
}

// CDNA5 async copy: global -> LDS, bypassing VGPRs. VDST operand = wave-relative
// LDS byte offset (low 32 bits of the generic pointer to a __shared__ object),
// VADDR = 64-bit global address ("v" on uint64_t prints as a VGPR pair).
__device__ inline void async_g2l_b128(const float* gp, float* lp) {
  const uint32_t lds_off = (uint32_t)(uintptr_t)lp;
  const uint64_t gaddr   = (uint64_t)(uintptr_t)gp;
  asm volatile("global_load_async_to_lds_b128 %0, %1, off"
               ::"v"(lds_off), "v"(gaddr)
               : "memory");
}

// Stage one KC=64-float chunk of x rows [b0..b0+15] for neuron n into LDS.
// 16 rows * 256B; each issue: 32 lanes * b128 covers 2 rows -> 8 issues.
__device__ inline void stage_chunk(const float* __restrict__ x, int b0, int n, int k,
                                   float* lbuf, int lane) {
  const int half = lane >> 4;        // which of the 2 rows this lane covers per issue
  const int col  = (lane & 15) * 4;  // float offset within the 64-float row chunk
#pragma unroll
  for (int i = 0; i < 8; ++i) {
    const int row = 2 * i + half;
    const float* gp = x + ((size_t)(b0 + row) * N_ + n) * K_ + k + col;
    float* lp = lbuf + row * ROWF + col;
    async_g2l_b128(gp, lp);
  }
}

// One WMMA k=32 step. A: 16 batches x 32 k from LDS (documented 16-bit A striping:
// lane group g in {0,1}, element e<8 -> K = 8g+e, e>=8 -> K = 16+8g+(e-8)).
// B: W[n, koff..koff+31] broadcast into all 16 columns via the mirrored striping,
// so every C column equals the v-vector for the 16 batches.
__device__ inline v8f wmma_step(const float* lbuf, const float* __restrict__ wrow,
                                int koff, int g, int m, v8f c) {
  const float* ra = lbuf + m * ROWF + koff + 8 * g;  // LDS (x row m)
  const float* rb = wrow + koff + 8 * g;             // global W (cache-hot, uniform per half-wave)
  v16h a, b;
#pragma unroll
  for (int j = 0; j < 8; ++j) {
    a[j]     = (_Float16)ra[j];
    a[j + 8] = (_Float16)ra[16 + j];
    b[j]     = (_Float16)rb[j];
    b[j + 8] = (_Float16)rb[16 + j];
  }
  return __builtin_amdgcn_wmma_f32_16x16x32_f16(
      /*neg_a=*/false, a, /*neg_b=*/false, b,
      /*c_mod=*/(short)0, c, /*reuse_a=*/false, /*reuse_b=*/false);
}

// One wave32 per (neuron n, 16-batch tile). grid = N_ * (B_/16) = 8192 blocks of 32.
__global__ __launch_bounds__(32) void mpjrd_fwd(const float* __restrict__ x,
                                                const float* __restrict__ W,
                                                const float* __restrict__ theta,
                                                float* __restrict__ spikes) {
  __shared__ float sb[2][16 * ROWF];  // double-buffered x stage (8.7 KB)

  const int bid  = blockIdx.x;
  const int n    = bid >> 3;          // consecutive blocks share n -> W reuse in cache
  const int b0   = (bid & 7) * 16;
  const int lane = (int)threadIdx.x;  // 0..31, full wave (EXEC all 1s for WMMA)
  const int g    = lane >> 4;
  const int m    = lane & 15;

  const float* wrow = W + (size_t)n * K_;
  __builtin_prefetch(wrow, 0, 0);  // global_prefetch_b8: warm the 2KB W row

  v8f c = {};

  stage_chunk(x, b0, n, 0, sb[0], lane);
  for (int it = 0; it < NCH; ++it) {
    if (it + 1 < NCH) stage_chunk(x, b0, n, (it + 1) * KC, sb[(it + 1) & 1], lane);
    if (it + 1 < NCH) wait_asynccnt<8>();  // chunk it done; chunk it+1 (8 issues) in flight
    else              wait_asynccnt<0>();
    const float* lb = sb[it & 1];
    const float* wr = wrow + it * KC;
    c = wmma_step(lb, wr, 0,  g, m, c);
    c = wmma_step(lb, wr, 32, g, m, c);
  }

  // C layout (broadcast B => all columns equal): lane group g holds batches 8g..8g+7
  // in c[0..7]. Lane (m == j) of each group writes batch 8g+j.
  const float th = theta[n];
#pragma unroll
  for (int j = 0; j < 8; ++j) {
    if (m == j) {
      const int mb = 8 * g + j;
      spikes[(size_t)(b0 + mb) * N_ + n] = (c[j] >= th) ? 1.0f : 0.0f;
    }
  }
}

// Per-neuron homeostatic update from the spike matrix (coalesced over n per b-step).
__global__ void mpjrd_finalize(const float* __restrict__ spikes,
                               const float* __restrict__ theta,
                               const float* __restrict__ r_hat,
                               float* __restrict__ out_tail) {  // [rates | thetas | r_hats]
  const int n = blockIdx.x * blockDim.x + threadIdx.x;
  if (n >= N_) return;
  float s = 0.0f;
  for (int b = 0; b < B_; ++b) s += spikes[(size_t)b * N_ + n];  // exact: sums of 0/1
  const float rate = s * (1.0f / (float)B_);
  const float rh   = 0.95f * r_hat[n] + 0.05f * rate;  // (1-ALPHA)*r_hat + ALPHA*rate
  const float th   = theta[n] + 0.1f * (rh - 0.1f);    // theta + THETA_LR*(rh - R_TARGET)
  out_tail[n]          = rate;  // rates
  out_tail[N_ + n]     = th;    // thetas
  out_tail[2 * N_ + n] = rh;    // r_hats
}

extern "C" void kernel_launch(void* const* d_in, const int* in_sizes, int n_in,
                              void* d_out, int out_size, void* d_ws, size_t ws_size,
                              hipStream_t stream) {
  (void)in_sizes; (void)n_in; (void)out_size; (void)d_ws; (void)ws_size;
  const float* x     = (const float*)d_in[0];  // [128,1024,8,64]
  const float* W     = (const float*)d_in[1];  // [1024,8,64]
  const float* theta = (const float*)d_in[2];  // [1024]
  const float* r_hat = (const float*)d_in[3];  // [1024]

  float* out    = (float*)d_out;
  float* spikes = out;                          // [128*1024]
  float* tail   = out + (size_t)B_ * N_;        // rates|thetas|r_hats, 3*1024

  mpjrd_fwd<<<dim3(N_ * (B_ / 16)), dim3(32), 0, stream>>>(x, W, theta, spikes);
  mpjrd_finalize<<<dim3((N_ + 255) / 256), dim3(256), 0, stream>>>(spikes, theta, r_hat, tail);
}
